// GPT2Model_40527311405243
// MI455X (gfx1250) — compile-verified, matrix operand order checked by reading, not run on hardware
//
#include <hip/hip_runtime.h>

// ---------------- constants (GPT-2: L=6, B=2, S=2048, D=768) ----------------
#define DD   768
#define SS   2048
#define BB   2
#define HH   12
#define HD   64
#define LL   6
#define VV   50257
#define NTOK (BB * SS)     // 4096
#define QKV3 (3 * DD)      // 2304
#define FF   (4 * DD)      // 3072

typedef __attribute__((ext_vector_type(16))) __bf16 v16bf;
typedef __attribute__((ext_vector_type(8)))  float  v8f;

union FragBF { v16bf v; uint4 q[2]; };
union Pack8  { uint4 q; __bf16 h[8]; };

// Load one 16x32 bf16 fragment (A-layout / transposed-B layout) from LDS.
// p points at &row[kstep*32 + kb], kb = (lane&16)?8:0; two 16-byte loads
// cover elements e=0..7 (K=kb+e) and e=8..15 (K=16+kb+e-8).
__device__ __forceinline__ v16bf load_frag(const __bf16* p) {
  FragBF f;
  f.q[0] = *reinterpret_cast<const uint4*>(p);
  f.q[1] = *reinterpret_cast<const uint4*>(p + 16);
  return f.v;
}

__device__ __forceinline__ v8f wmma_bf16(v16bf a, v16bf b, v8f c) {
  return __builtin_amdgcn_wmma_f32_16x16x32_bf16(false, a, false, b,
                                                 (short)0, c, false, false);
}

__device__ __forceinline__ v8f vzero8() {
  v8f z;
#pragma unroll
  for (int i = 0; i < 8; ++i) z[i] = 0.0f;
  return z;
}

// ---------------- embedding: x = wte[ids] + wpe[s] (f32) --------------------
__global__ __launch_bounds__(256) void embed_kernel(
    const int* __restrict__ ids, const float* __restrict__ wte,
    const float* __restrict__ wpe, float* __restrict__ x) {
  const int t  = blockIdx.x;
  const int s  = t & (SS - 1);
  const int id = ids[t];
#pragma unroll
  for (int e = 0; e < 3; ++e) {
    const int d = threadIdx.x + e * 256;
    x[(size_t)t * DD + d] = wte[(size_t)id * DD + d] + wpe[(size_t)s * DD + d];
  }
}

// ---------------- weight transpose + convert: W[K,N] f32 -> Wt[N,K] bf16 ----
__global__ __launch_bounds__(256) void transpose_cvt_kernel(
    const float* __restrict__ W, __bf16* __restrict__ Wt, int K, int N) {
  __shared__ float ls[32][33];
  const int n0 = blockIdx.x * 32;
  const int k0 = blockIdx.y * 32;
  const int tid = threadIdx.x;
#pragma unroll
  for (int e = 0; e < 4; ++e) {
    const int idx = tid + e * 256;
    const int rk = idx >> 5, cn = idx & 31;
    ls[rk][cn] = W[(size_t)(k0 + rk) * N + n0 + cn];
  }
  __syncthreads();
#pragma unroll
  for (int e = 0; e < 4; ++e) {
    const int idx = tid + e * 256;
    const int rn = idx >> 5, ck = idx & 31;
    Wt[(size_t)(n0 + rn) * K + k0 + ck] = (__bf16)ls[ck][rn];
  }
}

// ---------------- layernorm: f32 in, templated out (bf16 or f32) ------------
template <typename TOUT>
__global__ __launch_bounds__(256) void layernorm_kernel(
    const float* __restrict__ x, const float* __restrict__ g,
    const float* __restrict__ bta, TOUT* __restrict__ out) {
  __shared__ float red1[8];
  __shared__ float red2[8];
  const int t = blockIdx.x;
  const float* xr = x + (size_t)t * DD;
  float v[3];
  float s1 = 0.f, s2 = 0.f;
#pragma unroll
  for (int e = 0; e < 3; ++e) {
    v[e] = xr[threadIdx.x + e * 256];
    s1 += v[e];
    s2 += v[e] * v[e];
  }
#pragma unroll
  for (int off = 16; off >= 1; off >>= 1) {
    s1 += __shfl_xor(s1, off, 32);
    s2 += __shfl_xor(s2, off, 32);
  }
  const int w = threadIdx.x >> 5;
  if ((threadIdx.x & 31) == 0) { red1[w] = s1; red2[w] = s2; }
  __syncthreads();
  float t1 = 0.f, t2 = 0.f;
#pragma unroll
  for (int i = 0; i < 8; ++i) { t1 += red1[i]; t2 += red2[i]; }
  const float mu   = t1 * (1.0f / DD);
  const float var  = t2 * (1.0f / DD) - mu * mu;
  const float rstd = rsqrtf(var + 1e-5f);
#pragma unroll
  for (int e = 0; e < 3; ++e) {
    const int d = threadIdx.x + e * 256;
    out[(size_t)t * DD + d] = (TOUT)((v[e] - mu) * rstd * g[d] + bta[d]);
  }
}

// ---------------- bf16 WMMA GEMM: C = act(A @ Wt^T + bias) (+residual) ------
// A:[M,K] bf16, Wt:[N,K] bf16 (pre-transposed). Tile 128x256x64 per
// 256-thread block; 8 waves in 2(M)x4(N), each wave computes 64x64 of C
// (4x4 accumulators -> 16 WMMAs per 8 fragment loads per K-step).
// Next K-tile is prefetched (global_prefetch_b8) while this one computes.
#define GSTR 72   // LDS row stride in bf16 (144 B, 16B multiple)
template <int ACT, bool RES, typename TOUT>
__global__ __launch_bounds__(256) void wmma_gemm_kernel(
    const __bf16* __restrict__ A, const __bf16* __restrict__ Wt,
    const float* __restrict__ bias, const float* __restrict__ residual,
    TOUT* __restrict__ C, int M, int N, int K) {
  __shared__ __align__(16) __bf16 lsA[128][GSTR];   // [m][k]
  __shared__ __align__(16) __bf16 lsB[256][GSTR];   // [n][k]
  const int tid = threadIdx.x;
  const int n0  = blockIdx.x * 256;
  const int m0  = blockIdx.y * 128;
  const unsigned lane = tid & 31u;
  const int w   = tid >> 5;
  const int wm  = (w >> 2) * 64;          // 0 or 64
  const int wn  = (w & 3) * 64;           // 0..192
  const unsigned r  = lane & 15u;
  const unsigned kb = (lane & 16u) ? 8u : 0u;
  const int halfofs = (lane & 16u) ? 8 : 0;

  v8f acc[4][4];
#pragma unroll
  for (int mt = 0; mt < 4; ++mt)
#pragma unroll
    for (int nt = 0; nt < 4; ++nt) acc[mt][nt] = vzero8();

  const int kTiles = K >> 6;                 // BK = 64
  for (int kt = 0; kt < kTiles; ++kt) {
    __syncthreads();
    const int k0 = kt << 6;
#pragma unroll
    for (int e = 0; e < 4; ++e) {            // A: 128x64 bf16 = 1024 uint4
      const int u   = tid + e * 256;
      const int row = u >> 3;
      const int col = (u & 7) * 8;
      *reinterpret_cast<uint4*>(&lsA[row][col]) =
          *reinterpret_cast<const uint4*>(&A[(size_t)(m0 + row) * K + k0 + col]);
    }
#pragma unroll
    for (int e = 0; e < 8; ++e) {            // B: 256x64 bf16 = 2048 uint4
      const int u   = tid + e * 256;
      const int row = u >> 3;
      const int col = (u & 7) * 8;
      *reinterpret_cast<uint4*>(&lsB[row][col]) =
          *reinterpret_cast<const uint4*>(&Wt[(size_t)(n0 + row) * K + k0 + col]);
    }
    if (kt + 1 < kTiles) {                   // prefetch next K-tile rows
      const int k1 = k0 + 64;
      if (tid < 128) {
        __builtin_prefetch(&A[(size_t)(m0 + tid) * K + k1], 0, 3);
      } else {
        const int rr = (tid - 128) * 2;
        __builtin_prefetch(&Wt[(size_t)(n0 + rr) * K + k1], 0, 3);
        __builtin_prefetch(&Wt[(size_t)(n0 + rr + 1) * K + k1], 0, 3);
      }
    }
    __syncthreads();
#pragma unroll
    for (int ks = 0; ks < 2; ++ks) {
      v16bf af[4];
#pragma unroll
      for (int mt = 0; mt < 4; ++mt)
        af[mt] = load_frag(&lsA[wm + mt * 16 + r][ks * 32 + kb]);
#pragma unroll
      for (int nt = 0; nt < 4; ++nt) {
        const v16bf bfr = load_frag(&lsB[wn + nt * 16 + r][ks * 32 + kb]);
#pragma unroll
        for (int mt = 0; mt < 4; ++mt)
          acc[mt][nt] = wmma_bf16(af[mt], bfr, acc[mt][nt]);
      }
    }
  }

#pragma unroll
  for (int mt = 0; mt < 4; ++mt) {
#pragma unroll
    for (int nt = 0; nt < 4; ++nt) {
      const int col = n0 + wn + nt * 16 + (int)r;
      const float bv = bias[col];
#pragma unroll
      for (int i = 0; i < 8; ++i) {
        const int row = m0 + wm + mt * 16 + i + halfofs;
        float vv = acc[mt][nt][i] + bv;
        if (ACT == 1) vv = 0.5f * vv * (1.0f + erff(vv * 0.70710678118654752f));
        if (RES) vv += residual[(size_t)row * N + col];
        C[(size_t)row * N + col] = (TOUT)vv;
      }
    }
  }
}

// ---------------- fused causal flash attention (bf16 in / bf16 out) ---------
// grid (S/64, H, B); 128 threads = 4 waves, each wave owns 16 q rows.
#define ASTR 72
__global__ __launch_bounds__(128) void attn_kernel(
    const __bf16* __restrict__ qkv, __bf16* __restrict__ y) {
  __shared__ __align__(16) __bf16 Qs[64][ASTR];     // [q][hd]
  __shared__ __align__(16) __bf16 Ks[64][ASTR];     // [key][hd]
  __shared__ __align__(16) __bf16 Vt[64][ASTR];     // [hd][key]
  __shared__ __align__(16) __bf16 Ps[4][16][ASTR];  // per-wave P [q][key]

  const int qt = blockIdx.x;
  const int hh = blockIdx.y;
  const int b  = blockIdx.z;
  const int tid = threadIdx.x;
  const int w   = tid >> 5;
  const unsigned lane = tid & 31u;
  const unsigned r  = lane & 15u;
  const unsigned kb = (lane & 16u) ? 8u : 0u;
  const int halfofs = (lane & 16u) ? 8 : 0;

  const size_t rstride = QKV3;
  const size_t base = (size_t)b * SS * rstride;
  const int hofs = hh * HD;

#pragma unroll
  for (int e = 0; e < 4; ++e) {             // Q tile: 64x64 bf16 = 512 uint4
    const int u   = tid + e * 128;
    const int row = u >> 3;
    const int col = (u & 7) * 8;
    *reinterpret_cast<uint4*>(&Qs[row][col]) = *reinterpret_cast<const uint4*>(
        &qkv[base + (size_t)(qt * 64 + row) * rstride + hofs + col]);
  }
  __syncthreads();

  float mrun[8], lrun[8];
  v8f acc[4];
#pragma unroll
  for (int i = 0; i < 8; ++i) { mrun[i] = -1e30f; lrun[i] = 0.0f; }
#pragma unroll
  for (int n = 0; n < 4; ++n) acc[n] = vzero8();

  for (int kt = 0; kt <= qt; ++kt) {
    __syncthreads();
#pragma unroll
    for (int e = 0; e < 4; ++e) {
      const int u   = tid + e * 128;
      const int row = u >> 3;                // key index
      const int col = (u & 7) * 8;           // hd base
      const size_t tok = base + (size_t)(kt * 64 + row) * rstride;
      *reinterpret_cast<uint4*>(&Ks[row][col]) =
          *reinterpret_cast<const uint4*>(&qkv[tok + DD + hofs + col]);
      Pack8 pv;
      pv.q = *reinterpret_cast<const uint4*>(&qkv[tok + 2 * DD + hofs + col]);
#pragma unroll
      for (int j = 0; j < 8; ++j) Vt[col + j][row] = pv.h[j];
    }
    __syncthreads();

    // S = Q @ K^T
    v8f sc[4];
#pragma unroll
    for (int j = 0; j < 4; ++j) sc[j] = vzero8();
#pragma unroll
    for (int ks = 0; ks < 2; ++ks) {
      const v16bf aq = load_frag(&Qs[w * 16 + r][ks * 32 + kb]);
#pragma unroll
      for (int j = 0; j < 4; ++j) {
        const v16bf bk = load_frag(&Ks[j * 16 + r][ks * 32 + kb]);
        sc[j] = wmma_bf16(aq, bk, sc[j]);
      }
    }

    // scale + causal mask + online softmax
    const int q0 = qt * 64 + w * 16;
    const int k0 = kt * 64;
#pragma unroll
    for (int i = 0; i < 8; ++i) {
      const int rowg = q0 + i + halfofs;
      float mloc = -1e30f;
#pragma unroll
      for (int j = 0; j < 4; ++j) {
        const int colg = k0 + j * 16 + (int)r;
        float s = sc[j][i] * 0.125f;         // 1/sqrt(HD)
        if (colg > rowg) s = -1e30f;
        sc[j][i] = s;
        mloc = fmaxf(mloc, s);
      }
#pragma unroll
      for (int off = 1; off < 16; off <<= 1)
        mloc = fmaxf(mloc, __shfl_xor(mloc, off, 32));
      const float mnew  = fmaxf(mrun[i], mloc);
      const float scale = __expf(mrun[i] - mnew);
      float rsum = 0.0f;
#pragma unroll
      for (int j = 0; j < 4; ++j) {
        const float p = __expf(sc[j][i] - mnew);
        sc[j][i] = p;
        rsum += p;
      }
#pragma unroll
      for (int off = 1; off < 16; off <<= 1)
        rsum += __shfl_xor(rsum, off, 32);
      lrun[i] = lrun[i] * scale + rsum;
      mrun[i] = mnew;
#pragma unroll
      for (int n = 0; n < 4; ++n) acc[n][i] *= scale;
    }

    // C-layout -> A-layout reshape of P via wave-private LDS
#pragma unroll
    for (int j = 0; j < 4; ++j)
#pragma unroll
      for (int i = 0; i < 8; ++i)
        Ps[w][i + halfofs][j * 16 + r] = (__bf16)sc[j][i];

    // O += P @ V
#pragma unroll
    for (int ks = 0; ks < 2; ++ks) {
      const v16bf ap = load_frag(&Ps[w][r][ks * 32 + kb]);
#pragma unroll
      for (int n = 0; n < 4; ++n) {
        const v16bf bv = load_frag(&Vt[n * 16 + r][ks * 32 + kb]);
        acc[n] = wmma_bf16(ap, bv, acc[n]);
      }
    }
  }

#pragma unroll
  for (int i = 0; i < 8; ++i) {
    const float inv = 1.0f / lrun[i];
    const int qg = qt * 64 + w * 16 + i + halfofs;
    const size_t orow = ((size_t)b * SS + qg) * DD + hofs;
#pragma unroll
    for (int n = 0; n < 4; ++n)
      y[orow + n * 16 + r] = (__bf16)(acc[n][i] * inv);
  }
}

// ---------------- last-token logits (f32, bandwidth-optimal) ----------------
// wave-per-vocab-row, lane-strided coalesced reads of wte; both batch rows
// are dotted in the same pass so wte (154 MB) is streamed exactly once.
__global__ __launch_bounds__(256) void logits_kernel(
    const float* __restrict__ h, const float* __restrict__ wte,
    float* __restrict__ out) {
  __shared__ float xr0[DD];
  __shared__ float xr1[DD];
  const int tid = threadIdx.x;
#pragma unroll
  for (int e = 0; e < 3; ++e) {
    const int d = tid + e * 256;
    xr0[d] = h[((size_t)0 * SS + SS - 1) * DD + d];
    xr1[d] = h[((size_t)1 * SS + SS - 1) * DD + d];
  }
  __syncthreads();
  const int w    = tid >> 5;
  const int lane = tid & 31;
  const int v = blockIdx.x * 8 + w;
  if (v < VV) {
    const float* wr = wte + (size_t)v * DD;
    float s0 = 0.0f, s1 = 0.0f;
#pragma unroll
    for (int e = 0; e < DD / 32; ++e) {
      const int d = lane + e * 32;
      const float wv = wr[d];
      s0 += wv * xr0[d];
      s1 += wv * xr1[d];
    }
#pragma unroll
    for (int off = 16; off >= 1; off >>= 1) {
      s0 += __shfl_xor(s0, off, 32);
      s1 += __shfl_xor(s1, off, 32);
    }
    if (lane == 0) {
      out[v]      = s0;
      out[VV + v] = s1;
    }
  }
}

// ---------------- driver ----------------------------------------------------
extern "C" void kernel_launch(void* const* d_in, const int* in_sizes, int n_in,
                              void* d_out, int out_size, void* d_ws, size_t ws_size,
                              hipStream_t stream) {
  (void)in_sizes; (void)n_in; (void)out_size; (void)ws_size;
  const int*   ids         = (const int*)d_in[0];
  const float* wte         = (const float*)d_in[1];
  const float* wpe         = (const float*)d_in[2];
  const float* ln1_g       = (const float*)d_in[3];
  const float* ln1_b       = (const float*)d_in[4];
  const float* qkv_w       = (const float*)d_in[5];
  const float* qkv_b       = (const float*)d_in[6];
  const float* attn_proj_w = (const float*)d_in[7];
  const float* attn_proj_b = (const float*)d_in[8];
  const float* ln2_g       = (const float*)d_in[9];
  const float* ln2_b       = (const float*)d_in[10];
  const float* fc_w        = (const float*)d_in[11];
  const float* fc_b        = (const float*)d_in[12];
  const float* mlp_proj_w  = (const float*)d_in[13];
  const float* mlp_proj_b  = (const float*)d_in[14];
  const float* lnf_g       = (const float*)d_in[15];
  const float* lnf_b       = (const float*)d_in[16];
  float* out = (float*)d_out;

  // workspace layout
  char* p = (char*)d_ws;
  float*  x    = (float*)p;  p += (size_t)NTOK * DD * 4;     // residual stream
  float*  hf   = (float*)p;  p += (size_t)NTOK * DD * 4;     // final LN (f32)
  __bf16* hb   = (__bf16*)p; p += (size_t)NTOK * DD * 2;     // LN out (bf16)
  __bf16* qkvb = (__bf16*)p; p += (size_t)NTOK * QKV3 * 2;
  __bf16* ybuf = (__bf16*)p; p += (size_t)NTOK * DD * 2;
  __bf16* h2   = (__bf16*)p; p += (size_t)NTOK * FF * 2;
  __bf16* qkvwT = (__bf16*)p; p += (size_t)LL * QKV3 * DD * 2;  // [N,K]
  __bf16* apwT  = (__bf16*)p; p += (size_t)LL * DD * DD * 2;
  __bf16* fcwT  = (__bf16*)p; p += (size_t)LL * FF * DD * 2;
  __bf16* mpwT  = (__bf16*)p; p += (size_t)LL * DD * FF * 2;

  // one-time (per launch) weight transpose + bf16 convert
  for (int l = 0; l < LL; ++l) {
    transpose_cvt_kernel<<<dim3(QKV3 / 32, DD / 32), 256, 0, stream>>>(
        qkv_w + (size_t)l * DD * QKV3, qkvwT + (size_t)l * QKV3 * DD, DD, QKV3);
    transpose_cvt_kernel<<<dim3(DD / 32, DD / 32), 256, 0, stream>>>(
        attn_proj_w + (size_t)l * DD * DD, apwT + (size_t)l * DD * DD, DD, DD);
    transpose_cvt_kernel<<<dim3(FF / 32, DD / 32), 256, 0, stream>>>(
        fc_w + (size_t)l * DD * FF, fcwT + (size_t)l * FF * DD, DD, FF);
    transpose_cvt_kernel<<<dim3(DD / 32, FF / 32), 256, 0, stream>>>(
        mlp_proj_w + (size_t)l * FF * DD, mpwT + (size_t)l * DD * FF, FF, DD);
  }

  embed_kernel<<<NTOK, 256, 0, stream>>>(ids, wte, wpe, x);

  for (int l = 0; l < LL; ++l) {
    layernorm_kernel<__bf16><<<NTOK, 256, 0, stream>>>(
        x, ln1_g + l * DD, ln1_b + l * DD, hb);
    wmma_gemm_kernel<0, false, __bf16>
        <<<dim3(QKV3 / 256, NTOK / 128), 256, 0, stream>>>(
        hb, qkvwT + (size_t)l * QKV3 * DD, qkv_b + (size_t)l * QKV3,
        nullptr, qkvb, NTOK, QKV3, DD);
    attn_kernel<<<dim3(SS / 64, HH, BB), 128, 0, stream>>>(qkvb, ybuf);
    wmma_gemm_kernel<0, true, float>
        <<<dim3(DD / 256, NTOK / 128), 256, 0, stream>>>(
        ybuf, apwT + (size_t)l * DD * DD, attn_proj_b + (size_t)l * DD,
        x, x, NTOK, DD, DD);
    layernorm_kernel<__bf16><<<NTOK, 256, 0, stream>>>(
        x, ln2_g + l * DD, ln2_b + l * DD, hb);
    wmma_gemm_kernel<1, false, __bf16>
        <<<dim3(FF / 256, NTOK / 128), 256, 0, stream>>>(
        hb, fcwT + (size_t)l * FF * DD, fc_b + (size_t)l * FF,
        nullptr, h2, NTOK, FF, DD);
    wmma_gemm_kernel<0, true, float>
        <<<dim3(DD / 256, NTOK / 128), 256, 0, stream>>>(
        h2, mpwT + (size_t)l * DD * FF, mlp_proj_b + (size_t)l * DD,
        x, x, NTOK, DD, FF);
  }

  layernorm_kernel<float><<<NTOK, 256, 0, stream>>>(x, lnf_g, lnf_b, hf);
  logits_kernel<<<dim3((VV + 7) / 8, 1), 256, 0, stream>>>(hf, wte, out);
}
